// MLP_38663295598779
// MI455X (gfx1250) — matrix operand order repaired and checked by reading, not compiled, not run on hardware
//
#include <hip/hip_runtime.h>
#include <hip/hip_bf16.h>
#include <math.h>

#define T_SEQ 32768
#define NF    512
#define G4    2048          // 4 * NF gate rows
#define EPS   1e-5f
#define RNWG     32         // workgroups for recurrence (each owns 16 hidden units)
#define RTHREADS 512

typedef __bf16 bf16;
typedef __attribute__((ext_vector_type(16))) __bf16 v16bf;
typedef __attribute__((ext_vector_type(8)))  float  v8f;

// ---------------- elementwise f32 -> bf16 ----------------
__global__ void k_cvt_bf16(const float* __restrict__ in, bf16* __restrict__ out, long n) {
  long i = (long)blockIdx.x * blockDim.x + threadIdx.x;
  long stride = (long)gridDim.x * blockDim.x;
  for (; i < n; i += stride) out[i] = (bf16)in[i];
}

// ---------------- pad W3 [3,64] -> [16,64] bf16 (zero rows 3..15) ----------------
__global__ void k_pad_w3(const float* __restrict__ w3, bf16* __restrict__ out) {
  int i = blockIdx.x * blockDim.x + threadIdx.x;
  if (i < 16 * 64) {
    int r = i >> 6, k = i & 63;
    out[i] = (r < 3) ? (bf16)w3[r * 64 + k] : (bf16)0.0f;
  }
}

// ---------------- zero counters + BN-stats region (first 4KB of ws) ----------------
__global__ void k_init(unsigned* __restrict__ ws) {
  int i = blockIdx.x * blockDim.x + threadIdx.x;
  if (i < 1024) ws[i] = 0u;
}

// ---------------- WMMA GEMM: out[M,Nout] = A[M,K](bf16) @ W[N,K]^T(bf16) + bias ----------------
// One wave per 16x16 tile. Fragment layouts per CDNA5 ISA 7.12.2:
//   A 16x32 bf16: lane row = lane&15; elems 0..7 -> K = kb + (lane>=16?8:0) + 0..7
//                                     elems 8..15 -> K = kb+16 + (lane>=16?8:0) + 0..7
//   B 32x16 bf16: lane col = lane&15; same K mapping as A
//   C/D f32: acc[r] -> row = 16*tm + r + (lane>=16?8:0), col = 16*tn + (lane&15)
__global__ void __launch_bounds__(32) k_gemm_bf16(
    const bf16* __restrict__ A, const bf16* __restrict__ W,
    const float* __restrict__ b0, const float* __restrict__ b1,
    float* __restrict__ out, int M, int N, int K, int Nout, int relu) {
  const int lane = threadIdx.x;
  const int tn = blockIdx.x;
  const int tm = blockIdx.y;
  const int half = lane >> 4;
  const int l16 = lane & 15;
  const bf16* Ap = A + (long)(tm * 16 + l16) * K;
  const bf16* Wp = W + (long)(tn * 16 + l16) * K;
  v8f acc = {};
  for (int kb = 0; kb < K; kb += 32) {
    union { v16bf v; uint4 q[2]; } a, b;
    const int ka = kb + half * 8;
    a.q[0] = *(const uint4*)(Ap + ka);
    a.q[1] = *(const uint4*)(Ap + ka + 16);
    const int kw = kb + half * 8;
    b.q[0] = *(const uint4*)(Wp + kw);
    b.q[1] = *(const uint4*)(Wp + kw + 16);
    acc = __builtin_amdgcn_wmma_f32_16x16x32_bf16(
        false, a.v, false, b.v, (short)0, acc, false, false);
  }
  const int col = tn * 16 + l16;
  if (col < Nout) {
    float bias = 0.f;
    if (b0) bias += b0[col];
    if (b1) bias += b1[col];
#pragma unroll
    for (int r = 0; r < 8; ++r) {
      int row = tm * 16 + r + half * 8;
      float v = acc[r] + bias;
      if (relu) v = fmaxf(v, 0.f);
      out[(long)row * Nout + col] = v;
    }
  }
}

// ---------------- persistent LSTM recurrence ----------------
// 32 WGs; WG w owns hidden units [16w,16w+16) => 64 gate rows of w_hh resident in LDS
// (row-padded *513 for conflict-free access). Per step: 512 threads compute 64x512
// matvec partials, fold with xg, apply gates, publish 16 h values, global barrier,
// reload full h (agent-scope atomics keep L0 coherent).
__global__ void __launch_bounds__(RTHREADS) k_lstm_rec(
    const float* __restrict__ xg,   // [T, 2048]
    const float* __restrict__ w_hh, // [2048, 512]
    float* __restrict__ hs,         // [T, 512]  (output + h-exchange buffer)
    unsigned* __restrict__ ctr) {
  extern __shared__ float sm[];
  float* wlds = sm;                  // 64*513
  float* hsm  = wlds + 64 * 513;     // 512
  float* part = hsm + 512;           // 512
  float* gbuf = part + 512;          // 64
  const int tid = threadIdx.x;
  const int wg  = blockIdx.x;

  // stage this WG's 64 gate rows of w_hh into LDS (fp32)
  for (int idx = tid; idx < 64 * 512; idx += RTHREADS) {
    int r = idx >> 9, k = idx & 511;
    int grow = ((r >> 4) << 9) + wg * 16 + (r & 15); // gate*512 + unit
    wlds[r * 513 + k] = w_hh[(long)grow * 512 + k];
  }
  if (tid < 512) hsm[tid] = 0.f;
  float c = 0.f;                     // cell state, valid for tid < 16
  __syncthreads();

  const int j   = tid >> 6;          // k-chunk 0..7
  const int row = tid & 63;          // local gate row 0..63

  for (int t = 0; t < T_SEQ; ++t) {
    // 64-element partial dot (wave = 32 consecutive rows, same j -> conflict-free)
    const float* wr = wlds + row * 513 + j * 64;
    const float* hr = hsm + j * 64;
    float s = 0.f;
#pragma unroll
    for (int k = 0; k < 64; ++k) s += wr[k] * hr[k];
    part[row * 8 + j] = s;
    __syncthreads();

    if (tid < 64) {
      float g = xg[(long)t * G4 + ((tid >> 4) << 9) + wg * 16 + (tid & 15)];
#pragma unroll
      for (int jj = 0; jj < 8; ++jj) g += part[tid * 8 + jj];
      gbuf[tid] = g;
    }
    __syncthreads();

    if (tid < 16) {
      float gi = gbuf[tid], gf = gbuf[16 + tid], gc = gbuf[32 + tid], go = gbuf[48 + tid];
      float i_ = 1.f / (1.f + __expf(-gi));
      float f_ = 1.f / (1.f + __expf(-gf));
      float c_ = tanhf(gc);
      float o_ = 1.f / (1.f + __expf(-go));
      c = f_ * c + i_ * c_;
      float h = o_ * tanhf(c);
      __hip_atomic_store(&hs[(long)t * NF + wg * 16 + tid], h,
                         __ATOMIC_RELAXED, __HIP_MEMORY_SCOPE_AGENT);
    }
    __syncthreads();

    // device-wide step barrier: monotonic counter (zeroed per launch by k_init)
    if (tid == 0) {
      __hip_atomic_fetch_add(ctr, 1u, __ATOMIC_RELEASE, __HIP_MEMORY_SCOPE_AGENT);
      unsigned target = (unsigned)(t + 1) * RNWG;
      while (__hip_atomic_load(ctr, __ATOMIC_ACQUIRE, __HIP_MEMORY_SCOPE_AGENT) < target)
        __builtin_amdgcn_s_sleep(4);
    }
    __syncthreads();

    if (tid < 512)
      hsm[tid] = __hip_atomic_load(&hs[(long)t * NF + tid],
                                   __ATOMIC_RELAXED, __HIP_MEMORY_SCOPE_AGENT);
    __syncthreads();
  }
}

// ---------------- BatchNorm: per-column mean/var over T (biased) ----------------
__global__ void k_bn_stats(const float* __restrict__ a, int C, float* __restrict__ stats) {
  __shared__ float ssum[256], ssq[256];
  int c = blockIdx.x, tid = threadIdx.x;
  float s = 0.f, q = 0.f;
  for (int r = tid; r < T_SEQ; r += 256) {
    float v = a[(long)r * C + c];
    s += v; q += v * v;
  }
  ssum[tid] = s; ssq[tid] = q;
  __syncthreads();
  for (int o = 128; o > 0; o >>= 1) {
    if (tid < o) { ssum[tid] += ssum[tid + o]; ssq[tid] += ssq[tid + o]; }
    __syncthreads();
  }
  if (tid == 0) {
    float m = ssum[0] / (float)T_SEQ;
    stats[c] = m;
    stats[C + c] = ssq[0] / (float)T_SEQ - m * m;
  }
}

__global__ void k_bn_apply(float* __restrict__ a, int C, const float* __restrict__ stats,
                           const float* __restrict__ gamma, const float* __restrict__ beta) {
  long n = (long)T_SEQ * C;
  long i = (long)blockIdx.x * blockDim.x + threadIdx.x;
  long stride = (long)gridDim.x * blockDim.x;
  for (; i < n; i += stride) {
    int c = (int)(i % C);
    a[i] = (a[i] - stats[c]) * rsqrtf(stats[C + c] + EPS) * gamma[c] + beta[c];
  }
}

extern "C" void kernel_launch(void* const* d_in, const int* in_sizes, int n_in,
                              void* d_out, int out_size, void* d_ws, size_t ws_size,
                              hipStream_t stream) {
  const float* x     = (const float*)d_in[0];
  const float* w_ih0 = (const float*)d_in[1];
  const float* w_hh0 = (const float*)d_in[2];
  const float* b_ih0 = (const float*)d_in[3];
  const float* b_hh0 = (const float*)d_in[4];
  const float* w_ih1 = (const float*)d_in[5];
  const float* w_hh1 = (const float*)d_in[6];
  const float* b_ih1 = (const float*)d_in[7];
  const float* b_hh1 = (const float*)d_in[8];
  const float* w1 = (const float*)d_in[9];   const float* b1  = (const float*)d_in[10];
  const float* g1 = (const float*)d_in[11];  const float* be1 = (const float*)d_in[12];
  const float* w2 = (const float*)d_in[13];  const float* b2  = (const float*)d_in[14];
  const float* g2 = (const float*)d_in[15];  const float* be2 = (const float*)d_in[16];
  const float* w3 = (const float*)d_in[17];  const float* b3  = (const float*)d_in[18];

  char* ws = (char*)d_ws;
  unsigned* ctrs  = (unsigned*)ws;             // [0]=rec layer0, [1]=rec layer1
  float*    stats = (float*)(ws + 1024);       // <= 256 floats (mean|var)
  size_t off = 4096;
  float* xg  = (float*)(ws + off); off += (size_t)T_SEQ * G4 * 4;   // 256 MB
  float* hs0 = (float*)(ws + off); off += (size_t)T_SEQ * NF * 4;   //  64 MB
  float* hs1 = (float*)(ws + off); off += (size_t)T_SEQ * NF * 4;   //  64 MB
  float* a1  = (float*)(ws + off); off += (size_t)T_SEQ * 128 * 4;  //  16 MB
  float* a2  = (float*)(ws + off); off += (size_t)T_SEQ * 64 * 4;   //   8 MB
  bf16*  inb = (bf16*)(ws + off);  off += (size_t)T_SEQ * NF * 2;   //  32 MB (reused)
  bf16*  wb  = (bf16*)(ws + off);  off += (size_t)G4 * NF * 2;      //   2 MB (reused)

  const dim3 b32(32);
  const size_t recLds = (size_t)(64 * 513 + 512 + 512 + 64) * sizeof(float); // ~132.5 KB

  k_init<<<4, 256, 0, stream>>>(ctrs);

  // ----- LSTM layer 0 -----
  k_cvt_bf16<<<2048, 256, 0, stream>>>(x, inb, (long)T_SEQ * NF);
  k_cvt_bf16<<<2048, 256, 0, stream>>>(w_ih0, wb, (long)G4 * NF);
  k_gemm_bf16<<<dim3(G4 / 16, T_SEQ / 16), b32, 0, stream>>>(
      inb, wb, b_ih0, b_hh0, xg, T_SEQ, G4, NF, G4, 0);
  k_lstm_rec<<<RNWG, RTHREADS, recLds, stream>>>(xg, w_hh0, hs0, &ctrs[0]);

  // ----- LSTM layer 1 -----
  k_cvt_bf16<<<2048, 256, 0, stream>>>(hs0, inb, (long)T_SEQ * NF);
  k_cvt_bf16<<<2048, 256, 0, stream>>>(w_ih1, wb, (long)G4 * NF);
  k_gemm_bf16<<<dim3(G4 / 16, T_SEQ / 16), b32, 0, stream>>>(
      inb, wb, b_ih1, b_hh1, xg, T_SEQ, G4, NF, G4, 0);
  k_lstm_rec<<<RNWG, RTHREADS, recLds, stream>>>(xg, w_hh1, hs1, &ctrs[1]);

  // ----- fc1 + relu + BN -----
  k_cvt_bf16<<<2048, 256, 0, stream>>>(hs1, inb, (long)T_SEQ * NF);
  k_cvt_bf16<<<256, 256, 0, stream>>>(w1, wb, (long)128 * NF);
  k_gemm_bf16<<<dim3(128 / 16, T_SEQ / 16), b32, 0, stream>>>(
      inb, wb, b1, nullptr, a1, T_SEQ, 128, NF, 128, 1);
  k_bn_stats<<<128, 256, 0, stream>>>(a1, 128, stats);
  k_bn_apply<<<4096, 256, 0, stream>>>(a1, 128, stats, g1, be1);

  // ----- fc2 + relu + BN -----
  k_cvt_bf16<<<2048, 256, 0, stream>>>(a1, inb, (long)T_SEQ * 128);
  k_cvt_bf16<<<32, 256, 0, stream>>>(w2, wb, (long)64 * 128);
  k_gemm_bf16<<<dim3(64 / 16, T_SEQ / 16), b32, 0, stream>>>(
      inb, wb, b2, nullptr, a2, T_SEQ, 64, 128, 64, 1);
  k_bn_stats<<<64, 256, 0, stream>>>(a2, 64, stats);
  k_bn_apply<<<2048, 256, 0, stream>>>(a2, 64, stats, g2, be2);

  // ----- fc3 (N padded to 16, store only cols 0..2) -----
  k_cvt_bf16<<<2048, 256, 0, stream>>>(a2, inb, (long)T_SEQ * 64);
  k_pad_w3<<<4, 256, 0, stream>>>(w3, wb);
  k_gemm_bf16<<<dim3(1, T_SEQ / 16), b32, 0, stream>>>(
      inb, wb, b3, nullptr, (float*)d_out, T_SEQ, 16, 64, 3, 0);
}
